// UniMP_10703058502293
// MI455X (gfx1250) — matrix operand order repaired and checked by reading, not compiled, not run on hardware
//
#include <hip/hip_runtime.h>

// ---------------- problem constants (match reference) ----------------
#define NNODES 10000
#define NEDGES 320000
#define KDIM   256      // input dim of every layer's GEMMs (IN == HID == 256)
#define NHEADS 4
#define NTILES 4        // column tiles per wave in the WMMA GEMM (Nout % 64 == 0 always)

typedef __attribute__((ext_vector_type(16))) __bf16 v16bf;
typedef __attribute__((ext_vector_type(8)))  __bf16 v8bf;
typedef __attribute__((ext_vector_type(8)))  float  v8f;

// ---------------- small helpers ----------------
__device__ __forceinline__ float atomic_max_f32(float* addr, float val) {
  if (val >= 0.0f)
    return __int_as_float(atomicMax((int*)addr, __float_as_int(val)));
  return __uint_as_float(atomicMin((unsigned int*)addr, __float_as_uint(val)));
}

__global__ void fill_f32(float* __restrict__ p, float v, int n) {
  int i = blockIdx.x * blockDim.x + threadIdx.x;
  if (i < n) p[i] = v;
}

__global__ void cvt_bf16(const float* __restrict__ in, __bf16* __restrict__ out, int n) {
  int i = blockIdx.x * blockDim.x + threadIdx.x;
  if (i < n) out[i] = (__bf16)in[i];
}

// Wt[n][k] = (bf16) W[k][n]   (W is K x Nout row-major fp32)
__global__ void transpose_cvt_bf16(const float* __restrict__ W, __bf16* __restrict__ Wt,
                                   int K, int Nout) {
  int i = blockIdx.x * blockDim.x + threadIdx.x;
  if (i >= K * Nout) return;
  int n = i / K;
  int k = i - n * K;
  Wt[i] = (__bf16)W[(size_t)k * Nout + n];
}

// ---------------- WMMA GEMM: C[M x Nout] = A_bf16[M x 256] * Bt_bf16[Nout x 256]^T + bias ----
// One wave computes a 16x64 slab of C: one A fragment is reused across NTILES=4
// column tiles per K-step (2 A loads + 8 B loads feed 4 WMMAs), cutting A traffic 4x.
// 4 waves per block -> 256 output columns per block.
__global__ __launch_bounds__(128)
void gemm_bf16_wmma(const __bf16* __restrict__ A,   // [M x 256] row-major
                    const __bf16* __restrict__ Bt,  // [Nout x 256] row-major (W transposed)
                    const float*  __restrict__ bias,// [Nout]
                    float*        __restrict__ C,   // [M x Nout]
                    int Nout) {
  const int lane = threadIdx.x & 31;
  const int wave = threadIdx.x >> 5;
  const int mt = blockIdx.x;                         // row tile (16 rows)
  const int n0 = (blockIdx.y * 4 + wave) * (16 * NTILES); // first of 4 col tiles
  if (n0 >= Nout) return;                            // wave-uniform; EXEC stays full

  const bool hi = lane >= 16;
  const int  m  = mt * 16 + (lane & 15);

  // A layout (matches B): lane<16 holds K {k..k+15}; lane>=16 holds {k+16..k+31}
  const __bf16* arow = A + (size_t)m * KDIM + (hi ? 16 : 0);
  // B layout (matches A): lane<16 holds K {k..k+15} of column n; lane>=16 holds {k+16..k+31}
  const __bf16* brow[NTILES];
#pragma unroll
  for (int t = 0; t < NTILES; ++t)
    brow[t] = Bt + (size_t)(n0 + t * 16 + (lane & 15)) * KDIM + (hi ? 16 : 0);

  v8f acc[NTILES] = {};
#pragma unroll
  for (int k = 0; k < KDIM; k += 32) {
    v16bf a;
    *((v8bf*)&a)     = *(const v8bf*)(arow + k);
    *((v8bf*)&a + 1) = *(const v8bf*)(arow + k + 8);
#pragma unroll
    for (int t = 0; t < NTILES; ++t) {
      v16bf b;
      *((v8bf*)&b)     = *(const v8bf*)(brow[t] + k);
      *((v8bf*)&b + 1) = *(const v8bf*)(brow[t] + k + 8);
      acc[t] = __builtin_amdgcn_wmma_f32_16x16x32_bf16(false, a, false, b,
                                                       (short)0, acc[t], false, false);
    }
  }

  const int mrow = mt * 16 + (hi ? 8 : 0);  // VGPR r -> row M=r (lanes 0-15) / M=8+r (16-31)
#pragma unroll
  for (int t = 0; t < NTILES; ++t) {
    const int n = n0 + t * 16 + (lane & 15);
    const float bn = bias[n];
#pragma unroll
    for (int r = 0; r < 8; ++r)
      C[(size_t)(mrow + r) * Nout + n] = acc[t][r] + bn;
  }
}

// ---------------- edge attention ----------------
// scores[e,h] = dot(q[dst[e],h,:], k[src[e],h,:]) * scale
__global__ __launch_bounds__(256)
void edge_scores(const float* __restrict__ Q, const float* __restrict__ Kf,
                 const int* __restrict__ ei, float* __restrict__ sc,
                 int E, int dh, int hd, float scale) {
  int idx = blockIdx.x * blockDim.x + threadIdx.x;
  if (idx >= E * NHEADS) return;
  int e = idx >> 2, h = idx & 3;
  int s = ei[e], d = ei[E + e];
  const float* qp = Q  + (size_t)d * hd + h * dh;
  const float* kp = Kf + (size_t)s * hd + h * dh;
  float acc = 0.0f;
  for (int i = 0; i < dh; i += 4) {
    float4 q4 = *(const float4*)(qp + i);
    float4 k4 = *(const float4*)(kp + i);
    acc += q4.x * k4.x + q4.y * k4.y + q4.z * k4.z + q4.w * k4.w;
  }
  sc[idx] = acc * scale;
}

__global__ __launch_bounds__(256)
void seg_max(const float* __restrict__ sc, const int* __restrict__ ei,
             float* __restrict__ segm, int E) {
  int idx = blockIdx.x * blockDim.x + threadIdx.x;
  if (idx >= E * NHEADS) return;
  int e = idx >> 2, h = idx & 3;
  int d = ei[E + e];
  atomic_max_f32(&segm[d * NHEADS + h], sc[idx]);
}

__global__ __launch_bounds__(256)
void exp_sum(const float* __restrict__ sc, const float* __restrict__ segm,
             const int* __restrict__ ei, float* __restrict__ ex,
             float* __restrict__ segs, int E) {
  int idx = blockIdx.x * blockDim.x + threadIdx.x;
  if (idx >= E * NHEADS) return;
  int e = idx >> 2, h = idx & 3;
  int d = ei[E + e];
  float v = __expf(sc[idx] - segm[d * NHEADS + h]);
  ex[idx] = v;
  atomicAdd(&segs[d * NHEADS + h], v);
}

// out[dst, f] += v[src, f] * alpha[e, f/dh] ; one thread per (edge, feature)
__global__ __launch_bounds__(256)
void edge_agg(const float* __restrict__ V, const float* __restrict__ ex,
              const float* __restrict__ segs, const int* __restrict__ ei,
              float* __restrict__ Ob, int E, int hdShift, int dhShift) {
  int idx = blockIdx.x * blockDim.x + threadIdx.x;
  if (idx >= (E << hdShift)) return;
  int hd = 1 << hdShift;
  int e = idx >> hdShift;
  int f = idx & (hd - 1);
  int h = f >> dhShift;
  int s = ei[e], d = ei[E + e];
  float alpha = ex[(e << 2) + h] / segs[(d << 2) + h];
  atomicAdd(&Ob[((size_t)d << hdShift) + f], V[((size_t)s << hdShift) + f] * alpha);
}

// ---------------- gated residual (beta gate) ----------------
// out = concat ? Ob[n, 0:hd] : mean_h(Ob[n,h,:]) ; xr = S[n,:]
// g = sigmoid([out, xr, out - xr] . Wbeta) ; h = g*xr + (1-g)*out
__global__ __launch_bounds__(256)
void gated_residual(const float* __restrict__ Ob, const float* __restrict__ S,
                    const float* __restrict__ Wb, float* __restrict__ Hout,
                    int hd, int dh, int skip, int concat) {
  int n = blockIdx.x, t = threadIdx.x;
  __shared__ float red[256];
  __shared__ float gsh;
  float o = 0.0f, xr = 0.0f, p = 0.0f;
  if (t < skip) {
    if (concat) {
      o = Ob[(size_t)n * hd + t];
    } else {
      float a = 0.0f;
      for (int h = 0; h < NHEADS; ++h) a += Ob[(size_t)n * hd + h * dh + t];
      o = a * (1.0f / NHEADS);
    }
    xr = S[(size_t)n * skip + t];
    p = Wb[t] * o + Wb[skip + t] * xr + Wb[2 * skip + t] * (o - xr);
  }
  red[t] = p;
  __syncthreads();
  for (int s2 = 128; s2 > 0; s2 >>= 1) {
    if (t < s2) red[t] += red[t + s2];
    __syncthreads();
  }
  if (t == 0) gsh = 1.0f / (1.0f + __expf(-red[0]));
  __syncthreads();
  if (t < skip) Hout[(size_t)n * skip + t] = gsh * xr + (1.0f - gsh) * o;
}

// ---------------- layernorm + relu (D = 256, in-place safe) ----------------
__global__ __launch_bounds__(256)
void layernorm_relu(const float* __restrict__ X, const float* __restrict__ g,
                    const float* __restrict__ b, float* __restrict__ Y) {
  const int D = 256;
  int n = blockIdx.x, t = threadIdx.x;
  __shared__ float red[256];
  __shared__ float mu_s, rs_s;
  float x = X[(size_t)n * D + t];
  red[t] = x;
  __syncthreads();
  for (int s2 = 128; s2 > 0; s2 >>= 1) { if (t < s2) red[t] += red[t + s2]; __syncthreads(); }
  if (t == 0) mu_s = red[0] * (1.0f / D);
  __syncthreads();
  float dx = x - mu_s;
  red[t] = dx * dx;
  __syncthreads();
  for (int s2 = 128; s2 > 0; s2 >>= 1) { if (t < s2) red[t] += red[t + s2]; __syncthreads(); }
  if (t == 0) rs_s = rsqrtf(red[0] * (1.0f / D) + 1e-5f);
  __syncthreads();
  float y = dx * rs_s * g[t] + b[t];
  Y[(size_t)n * D + t] = fmaxf(y, 0.0f);
}

// ---------------- host-side layer driver ----------------
static void run_layer(const float* hin,
                      const float* Wq, const float* bq, const float* Wk, const float* bk,
                      const float* Wv, const float* bv, const float* Ws, const float* bs,
                      const float* Wbeta, const int* ei, float* hout,
                      int hd, int dh, int skip, int concat,
                      __bf16* xb, __bf16* Wt, float* Q, float* K, float* V, float* S,
                      float* Ob, float* sc, float* ex, float* segm, float* segs,
                      hipStream_t stream) {
  const int N = NNODES, E = NEDGES;
  // activations -> bf16
  cvt_bf16<<<(N * KDIM + 255) / 256, 256, 0, stream>>>(hin, xb, N * KDIM);

  // Q, K, V projections (transpose-convert weight, then WMMA GEMM; Wt reused serially).
  // Each block covers 256 output columns (4 waves x 4 tiles x 16).
  dim3 gq(N / 16, (hd + 255) / 256);
  transpose_cvt_bf16<<<(KDIM * hd + 255) / 256, 256, 0, stream>>>(Wq, Wt, KDIM, hd);
  gemm_bf16_wmma<<<gq, 128, 0, stream>>>(xb, Wt, bq, Q, hd);
  transpose_cvt_bf16<<<(KDIM * hd + 255) / 256, 256, 0, stream>>>(Wk, Wt, KDIM, hd);
  gemm_bf16_wmma<<<gq, 128, 0, stream>>>(xb, Wt, bk, K, hd);
  transpose_cvt_bf16<<<(KDIM * hd + 255) / 256, 256, 0, stream>>>(Wv, Wt, KDIM, hd);
  gemm_bf16_wmma<<<gq, 128, 0, stream>>>(xb, Wt, bv, V, hd);
  // skip projection
  dim3 gs(N / 16, (skip + 255) / 256);
  transpose_cvt_bf16<<<(KDIM * skip + 255) / 256, 256, 0, stream>>>(Ws, Wt, KDIM, skip);
  gemm_bf16_wmma<<<gs, 128, 0, stream>>>(xb, Wt, bs, S, skip);

  // attention: init accumulators
  fill_f32<<<(N * NHEADS + 255) / 256, 256, 0, stream>>>(segm, -INFINITY, N * NHEADS);
  fill_f32<<<(N * NHEADS + 255) / 256, 256, 0, stream>>>(segs, 0.0f, N * NHEADS);
  fill_f32<<<(N * hd + 255) / 256, 256, 0, stream>>>(Ob, 0.0f, N * hd);

  const float scale = 1.0f / sqrtf((float)dh);
  const int nEH = E * NHEADS;
  edge_scores<<<(nEH + 255) / 256, 256, 0, stream>>>(Q, K, ei, sc, E, dh, hd, scale);
  seg_max<<<(nEH + 255) / 256, 256, 0, stream>>>(sc, ei, segm, E);
  exp_sum<<<(nEH + 255) / 256, 256, 0, stream>>>(sc, segm, ei, ex, segs, E);

  const int hdShift = (hd == 512) ? 9 : (hd == 256 ? 8 : 7);
  const int dhShift = (dh == 128) ? 7 : 6;
  const long long nEF = (long long)E << hdShift;
  edge_agg<<<(unsigned)((nEF + 255) / 256), 256, 0, stream>>>(V, ex, segs, ei, Ob, E, hdShift, dhShift);

  gated_residual<<<N, 256, 0, stream>>>(Ob, S, Wbeta, hout, hd, dh, skip, concat);
}

// ---------------- entry point ----------------
// d_in layout (dict insertion order, recursive):
//   [0] x (N*256 f32)
//   layer l in 0..2, base = 1 + 9*l:
//     +0 Wq  +1 bq  +2 Wk  +3 bk  +4 Wv  +5 bv  +6 Ws  +7 bs  +8 Wbeta
//   [28] ln0.g  [29] ln0.b  [30] ln1.g  [31] ln1.b
//   [32] edge_index (2*E int)
extern "C" void kernel_launch(void* const* d_in, const int* in_sizes, int n_in,
                              void* d_out, int out_size, void* d_ws, size_t ws_size,
                              hipStream_t stream) {
  (void)in_sizes; (void)n_in; (void)out_size; (void)ws_size;
  const float* x = (const float*)d_in[0];
  auto P = [&](int l, int j) { return (const float*)d_in[1 + l * 9 + j]; };
  const float* lng0 = (const float*)d_in[28];
  const float* lnb0 = (const float*)d_in[29];
  const float* lng1 = (const float*)d_in[30];
  const float* lnb1 = (const float*)d_in[31];
  const int*   ei   = (const int*)d_in[32];

  // carve workspace (~129 MB)
  char* w = (char*)d_ws;
  auto carve = [&](size_t bytes) { void* p = w; w += (bytes + 255) & ~(size_t)255; return p; };
  float*  hA   = (float*) carve((size_t)NNODES * 256 * 4);
  float*  hB   = (float*) carve((size_t)NNODES * 256 * 4);
  float*  Q    = (float*) carve((size_t)NNODES * 512 * 4);
  float*  K    = (float*) carve((size_t)NNODES * 512 * 4);
  float*  V    = (float*) carve((size_t)NNODES * 512 * 4);
  float*  S    = (float*) carve((size_t)NNODES * 256 * 4);
  float*  Ob   = (float*) carve((size_t)NNODES * 512 * 4);
  float*  sc   = (float*) carve((size_t)NEDGES * NHEADS * 4);
  float*  ex   = (float*) carve((size_t)NEDGES * NHEADS * 4);
  float*  segm = (float*) carve((size_t)NNODES * NHEADS * 4);
  float*  segs = (float*) carve((size_t)NNODES * NHEADS * 4);
  __bf16* xb   = (__bf16*)carve((size_t)NNODES * 256 * 2);
  __bf16* Wt   = (__bf16*)carve((size_t)512 * 256 * 2);

  // layer 0: din=256, hd=256, dh=64, skip=256, concat
  run_layer(x,  P(0,0), P(0,1), P(0,2), P(0,3), P(0,4), P(0,5), P(0,6), P(0,7), P(0,8),
            ei, hA, 256, 64, 256, 1, xb, Wt, Q, K, V, S, Ob, sc, ex, segm, segs, stream);
  layernorm_relu<<<NNODES, 256, 0, stream>>>(hA, lng0, lnb0, hA);

  // layer 1: same shapes
  run_layer(hA, P(1,0), P(1,1), P(1,2), P(1,3), P(1,4), P(1,5), P(1,6), P(1,7), P(1,8),
            ei, hB, 256, 64, 256, 1, xb, Wt, Q, K, V, S, Ob, sc, ex, segm, segs, stream);
  layernorm_relu<<<NNODES, 256, 0, stream>>>(hB, lng1, lnb1, hB);

  // layer 2: hd=512, dh=128, skip=128, mean over heads; writes final output
  run_layer(hB, P(2,0), P(2,1), P(2,2), P(2,3), P(2,4), P(2,5), P(2,6), P(2,7), P(2,8),
            ei, (float*)d_out, 512, 128, 128, 0, xb, Wt, Q, K, V, S, Ob, sc, ex, segm, segs, stream);
}
